// GEFM_5102421147993
// MI455X (gfx1250) — matrix operand order repaired and checked by reading, compiled, not run on hardware
//
#include <hip/hip_runtime.h>

// ---------------------------------------------------------------------------
// CDNA5 (gfx1250, wave32) implementation of the dual cross-attention block.
// All GEMM-like work (6 convs + 2 flash-attentions) runs on
// v_wmma_f32_16x16x32_f16 with fp32 accumulators. Attention operand
// streaming uses GLOBAL_LOAD_ASYNC_TO_LDS_B128 (ASYNCcnt path).
// ---------------------------------------------------------------------------

typedef _Float16 v8h  __attribute__((ext_vector_type(8)));
typedef _Float16 v16h __attribute__((ext_vector_type(16)));
typedef float    v8f  __attribute__((ext_vector_type(8)));

__device__ __forceinline__ v8f zero8() {
    v8f z = {0.f, 0.f, 0.f, 0.f, 0.f, 0.f, 0.f, 0.f};
    return z;
}

__device__ __forceinline__ v16h pack16(v8h lo, v8h hi) {
    v16h r;
#pragma unroll
    for (int i = 0; i < 8; ++i) { r[i] = lo[i]; r[i + 8] = hi[i]; }
    return r;
}

// B-matrix fragment (32x16, 16-bit): lane (l&15) = column n; K rows
// (l<16 ? 0..15 : 16..31) are 16 contiguous halves in K-major storage.
// `base` must already include the (lo?0:16) K offset.
__device__ __forceinline__ v16h load_bfrag(const _Float16* base) {
    v8h a = *(const v8h*)base;
    v8h b = *(const v8h*)(base + 8);
    return pack16(a, b);
}

// A-matrix fragment (16x32, 16-bit): lane (l&15) = row M.
// lanes 0-15: K0..15 ; lanes 16-31: K16..31 (contiguous, matches load_bfrag).
// `base` points at K=0 of the 32-wide K chunk (K contiguous).
__device__ __forceinline__ v16h load_afrag(const _Float16* base, bool lo) {
    const int o = lo ? 0 : 16;
    v8h a = *(const v8h*)(base + o);
    v8h b = *(const v8h*)(base + 8 + o);
    return pack16(a, b);
}

__device__ __forceinline__ v8f wmma16(v16h a, v16h b, v8f c) {
    return __builtin_amdgcn_wmma_f32_16x16x32_f16(
        /*neg_a=*/false, a, /*neg_b=*/false, b,
        /*c_mod=*/(short)0, c, /*reuse_a=*/false, /*reuse_b=*/false);
}

// Direct global -> LDS async copy (16B per lane), tracked by ASYNCcnt.
// Per the ISA flat-aperture rule the low 32 bits of a generic LDS pointer
// are the LDS byte address, which is what the VDST operand wants.
__device__ __forceinline__ void async_ld16(void* lds, const void* gptr) {
    const unsigned l = (unsigned)(unsigned long long)lds;
    asm volatile("global_load_async_to_lds_b128 %0, %1, off"
                 :: "v"(l), "v"(gptr)
                 : "memory");
}
__device__ __forceinline__ void wait_async0() {
    asm volatile("s_wait_asynccnt 0" ::: "memory");
}

// ---------------------------------------------------------------------------
// Weight packing: f32 OIHW -> f16 laid out in WMMA A-fragment order:
//   wf[(((kc*3+dy)*3+dx) * 64 + co) * 32 + ci_off] = w[co][kc*32+ci_off][dy][dx]
// so the conv inner loop builds an A-fragment with two 16B loads.
// ---------------------------------------------------------------------------
template <int CIN>
__global__ __launch_bounds__(256) void pack_weights(
    const float* __restrict__ w, _Float16* __restrict__ wf) {
    const int i = blockIdx.x * 256 + threadIdx.x;
    if (i >= CIN * 9 * 64) return;
    const int ci_off = i & 31;
    const int co     = (i >> 5) & 63;
    const int f      = i >> 11;          // kc*9 + dy*3 + dx
    const int dx     = f % 3;
    const int dy     = (f / 3) % 3;
    const int kc     = f / 9;
    const int ci     = kc * 32 + ci_off;
    wf[i] = (_Float16)w[(((size_t)co * CIN + ci) * 3 + dy) * 3 + dx];
}

// ---------------------------------------------------------------------------
// Conv3x3 (pad 1, stride 1) + folded-BN + ReLU as 9 shifted GEMMs on WMMA.
// One block (4 waves, 128 thr) computes one image row: 64 Cout x 64 px.
// SMODE: 0 = f16 transposed [b][n][64]  (for Q / K operands of attention)
//        1 = f16 normal     [b][64][n]  (for V operands)
//        2 = f32 normal     [b][64][n]  (final output)
// ---------------------------------------------------------------------------
template <int CIN, int SMODE, bool CONCAT, bool F16IN>
__global__ __launch_bounds__(128) void conv3_wmma(
    const float* __restrict__ inA, const float* __restrict__ inB,
    const _Float16* __restrict__ inH, const _Float16* __restrict__ wf,
    const float* __restrict__ sc, const float* __restrict__ bi,
    _Float16* __restrict__ o16, float* __restrict__ o32) {
    constexpr int CINP = CIN + 8;  // pad stride -> bank-conflict-free ds reads
    __shared__ _Float16 tile[3][66][CINP];

    const int tid = threadIdx.x;
    const int b   = blockIdx.x >> 6;
    const int h   = blockIdx.x & 63;

    // Stage 3 zero-padded input rows, converted to f16, channels-last.
    for (int i = tid; i < 3 * 66 * CIN; i += 128) {
        const int px  = i % 66;
        const int r2  = i / 66;
        const int c   = r2 % CIN;
        const int dy  = r2 / CIN;
        const int row = h + dy - 1;
        float v = 0.f;
        if (px >= 1 && px <= 64 && row >= 0 && row < 64) {
            const int xx = px - 1;
            if constexpr (F16IN) {
                v = (float)inH[(((size_t)b * CIN + c) * 64 + row) * 64 + xx];
            } else if constexpr (CONCAT) {
                v = (c < 64)
                    ? inA[(((size_t)b * 64 + c) * 64 + row) * 64 + xx]
                    : inB[(((size_t)b * 64 + (c - 64)) * 64 + row) * 64 + xx];
            } else {
                v = inA[(((size_t)b * 64 + c) * 64 + row) * 64 + xx];
            }
        }
        tile[dy][px][c] = (_Float16)v;
    }
    __syncthreads();

    const int  wave = tid >> 5;
    const int  lane = tid & 31;
    const int  m    = lane & 15;
    const bool lo   = lane < 16;

    v8f acc[4];
#pragma unroll
    for (int nt = 0; nt < 4; ++nt) acc[nt] = zero8();

    for (int kc = 0; kc < CIN / 32; ++kc)
        for (int dy = 0; dy < 3; ++dy)
            for (int dx = 0; dx < 3; ++dx) {
                // Pre-packed weight A-fragment: two 16B loads.
                const int f = (kc * 3 + dy) * 3 + dx;
                const _Float16* ab =
                    wf + ((size_t)f * 64 + wave * 16 + m) * 32;
                v16h a = load_afrag(ab, lo);
#pragma unroll
                for (int nt = 0; nt < 4; ++nt) {
                    const _Float16* bp =
                        &tile[dy][nt * 16 + m + dx][kc * 32 + (lo ? 0 : 16)];
                    acc[nt] = wmma16(a, load_bfrag(bp), acc[nt]);
                }
            }

    // Epilogue: BN affine + ReLU, store in requested layout.
#pragma unroll
    for (int nt = 0; nt < 4; ++nt) {
        const int    px  = nt * 16 + m;
        const size_t pxg = (size_t)b * 4096 + h * 64 + px;
        if constexpr (SMODE == 0) {
            v8h pk;
#pragma unroll
            for (int r = 0; r < 8; ++r) {
                const int co = wave * 16 + (lo ? 0 : 8) + r;
                float v = fmaxf(acc[nt][r] * sc[co] + bi[co], 0.f);
                pk[r] = (_Float16)v;
            }
            *(v8h*)(o16 + pxg * 64 + wave * 16 + (lo ? 0 : 8)) = pk;
        } else {
#pragma unroll
            for (int r = 0; r < 8; ++r) {
                const int co = wave * 16 + (lo ? 0 : 8) + r;
                float v = fmaxf(acc[nt][r] * sc[co] + bi[co], 0.f);
                if constexpr (SMODE == 1)
                    o16[((size_t)b * 64 + co) * 4096 + h * 64 + px] = (_Float16)v;
                else
                    o32[((size_t)b * 64 + co) * 4096 + h * 64 + px] = v;
            }
        }
    }
}

// ---------------------------------------------------------------------------
// Flash attention per the reference semantics:
//   out[c,m] = (1/l_m) * sum_n exp(S[m,n]-max_m) * V[c,n],
//   S[m,n]   = sum_c K[c,m] * Q[c,n]   (softmax axis = Q positions!)
// rowsT = K^T  [b][m][c]  (A operand, register resident per wave)
// colsT = Q^T  [b][n][c]  (B operand, streamed via async-to-LDS)
// Vv    = V    [b][c][n]
// One block = 4 waves = 64 output positions; stream n in 64-wide chunks.
// refine[ch] = gamma*attn + resid, written f16 into the concat buffer.
// ---------------------------------------------------------------------------
__global__ __launch_bounds__(128) void attend_wmma(
    const _Float16* __restrict__ rowsT, const _Float16* __restrict__ Vv,
    const _Float16* __restrict__ colsT, const float* __restrict__ resid,
    const float* __restrict__ gamma, _Float16* __restrict__ refine,
    int chbase) {
    __shared__ _Float16 Qs[64][72];      // streamed Q chunk, [n][c]
    __shared__ _Float16 Vs[64][72];      // streamed V chunk, [c][n]
    __shared__ _Float16 Ps[4][16][72];   // per-wave P (m x n), D->B relayout
    __shared__ float    red[4][16];      // per-wave alpha / 1/l broadcast

    const int  tid  = threadIdx.x;
    const int  wave = tid >> 5;
    const int  lane = tid & 31;
    const int  m    = lane & 15;
    const bool lo   = lane < 16;
    const int  b    = blockIdx.y;
    const int  qm   = blockIdx.x * 64 + wave * 16 + m;  // this lane's out pos

    // A-fragments: K^T rows for this wave (fixed all loop).
    const _Float16* krow = rowsT + ((size_t)b * 4096 + qm) * 64;
    v16h ka[2];
#pragma unroll
    for (int kc = 0; kc < 2; ++kc) ka[kc] = load_afrag(krow + kc * 32, lo);

    v8f O[4];
#pragma unroll
    for (int ct = 0; ct < 4; ++ct) O[ct] = zero8();
    float mrow[8], lrow[8];
#pragma unroll
    for (int r = 0; r < 8; ++r) { mrow[r] = -1e30f; lrow[r] = 0.f; }

    for (int nb = 0; nb < 64; ++nb) {
        const int n0 = nb * 64;
        // Stage Q chunk ([n][c]) and V chunk ([c][n]) straight into LDS via
        // the CDNA5 async path (no VGPR round-trip); per-lane addresses
        // absorb the 64 -> 72 row-stride re-pad.
        for (int i = tid; i < 512; i += 128) {
            const int row = i >> 3, g = (i & 7) * 8;
            async_ld16(&Qs[row][g],
                       colsT + ((size_t)b * 4096 + n0 + row) * 64 + g);
            async_ld16(&Vs[row][g],
                       Vv + ((size_t)b * 64 + row) * 4096 + n0 + g);
        }
        wait_async0();
        __syncthreads();

        // GEMM1: S[m, n0..n0+63] = K^T x Q
        v8f S[4];
#pragma unroll
        for (int nt = 0; nt < 4; ++nt) {
            S[nt] = zero8();
#pragma unroll
            for (int kc = 0; kc < 2; ++kc) {
                v16h bf = load_bfrag(&Qs[nt * 16 + m][kc * 32 + (lo ? 0 : 16)]);
                S[nt] = wmma16(ka[kc], bf, S[nt]);
            }
        }

        // Online softmax: row r of D-layout lives in one VGPR across a
        // 16-lane half -> reduce with xor-shuffles inside the half.
        float alpha[8], rsum[8];
#pragma unroll
        for (int r = 0; r < 8; ++r) {
            float cm = fmaxf(fmaxf(S[0][r], S[1][r]), fmaxf(S[2][r], S[3][r]));
            for (int off = 1; off < 16; off <<= 1)
                cm = fmaxf(cm, __shfl_xor(cm, off, 32));
            const float nm = fmaxf(mrow[r], cm);
            alpha[r] = __expf(mrow[r] - nm);
            mrow[r]  = nm;
            rsum[r]  = 0.f;
        }
#pragma unroll
        for (int nt = 0; nt < 4; ++nt)
#pragma unroll
            for (int r = 0; r < 8; ++r) {
                const float p = __expf(S[nt][r] - mrow[r]);
                rsum[r] += p;
                Ps[wave][r + (lo ? 0 : 8)][nt * 16 + m] = (_Float16)p;
            }
#pragma unroll
        for (int r = 0; r < 8; ++r) {
            float s = rsum[r];
            for (int off = 1; off < 16; off <<= 1) s += __shfl_xor(s, off, 32);
            lrow[r] = lrow[r] * alpha[r] + s;
        }
        if (lane == 0) {
#pragma unroll
            for (int r = 0; r < 8; ++r) red[wave][r] = alpha[r];
        }
        if (lane == 16) {
#pragma unroll
            for (int r = 0; r < 8; ++r) red[wave][8 + r] = alpha[r];
        }
        __syncthreads();
        const float am = red[wave][m];  // alpha for this lane's query m
#pragma unroll
        for (int ct = 0; ct < 4; ++ct)
#pragma unroll
            for (int r = 0; r < 8; ++r) O[ct][r] *= am;

        // GEMM2: O[c, m] += V x P^T
#pragma unroll
        for (int ct = 0; ct < 4; ++ct)
#pragma unroll
            for (int kn = 0; kn < 2; ++kn) {
                v16h va = load_afrag(&Vs[ct * 16 + m][kn * 32], lo);
                v16h pb = load_bfrag(&Ps[wave][m][kn * 32 + (lo ? 0 : 16)]);
                O[ct] = wmma16(va, pb, O[ct]);
            }
        __syncthreads();
    }

    // Final 1/l, gamma scale, residual add, f16 store into concat buffer.
    if (lane == 0) {
#pragma unroll
        for (int r = 0; r < 8; ++r) red[wave][r] = 1.f / lrow[r];
    }
    if (lane == 16) {
#pragma unroll
        for (int r = 0; r < 8; ++r) red[wave][8 + r] = 1.f / lrow[r];
    }
    __syncthreads();
    const float li = red[wave][m];
    const float g  = gamma[0];
#pragma unroll
    for (int ct = 0; ct < 4; ++ct)
#pragma unroll
        for (int r = 0; r < 8; ++r) {
            const int c = ct * 16 + (lo ? 0 : 8) + r;
            const float val =
                g * O[ct][r] * li + resid[((size_t)b * 64 + c) * 4096 + qm];
            refine[((size_t)b * 128 + chbase + c) * 4096 + qm] = (_Float16)val;
        }
}

// ---------------------------------------------------------------------------
extern "C" void kernel_launch(void* const* d_in, const int* in_sizes, int n_in,
                              void* d_out, int out_size, void* d_ws,
                              size_t ws_size, hipStream_t stream) {
    (void)in_sizes; (void)n_in; (void)out_size; (void)ws_size;
    const float* x    = (const float*)d_in[0];
    const float* y    = (const float*)d_in[1];
    const float* q_w  = (const float*)d_in[2];
    const float* q_s  = (const float*)d_in[3];
    const float* q_b  = (const float*)d_in[4];
    const float* rk_w = (const float*)d_in[5];
    const float* rk_s = (const float*)d_in[6];
    const float* rk_b = (const float*)d_in[7];
    const float* rv_w = (const float*)d_in[8];
    const float* rv_s = (const float*)d_in[9];
    const float* rv_b = (const float*)d_in[10];
    const float* ik_w = (const float*)d_in[11];
    const float* ik_s = (const float*)d_in[12];
    const float* ik_b = (const float*)d_in[13];
    const float* iv_w = (const float*)d_in[14];
    const float* iv_s = (const float*)d_in[15];
    const float* iv_b = (const float*)d_in[16];
    const float* sr_w = (const float*)d_in[17];
    const float* sr_s = (const float*)d_in[18];
    const float* sr_b = (const float*)d_in[19];
    const float* g1   = (const float*)d_in[20];
    const float* g2   = (const float*)d_in[21];

    // Workspace layout (all 16B aligned). NBC bytes = half of one
    // [B][4096][64] f16 buffer; each activation buffer is 2*NBC bytes.
    const size_t NBC = (size_t)4 * 4096 * 64;
    char* ws = (char*)d_ws;
    _Float16* Q16  = (_Float16*)(ws);             // Q^T    [b][n][64]
    _Float16* RK16 = (_Float16*)(ws + 2 * NBC);   // rgbK^T [b][n][64]
    _Float16* RV16 = (_Float16*)(ws + 4 * NBC);   // rgbV   [b][64][n]
    _Float16* IK16 = (_Float16*)(ws + 6 * NBC);   // infK^T
    _Float16* IV16 = (_Float16*)(ws + 8 * NBC);   // infV
    _Float16* REF  = (_Float16*)(ws + 10 * NBC);  // concat refine [b][128][n]
    // Packed f16 weights (A-fragment order).
    const size_t W64  = (size_t)64 * 9 * 64;      // elements, CIN=64
    const size_t W128 = (size_t)128 * 9 * 64;     // elements, CIN=128
    _Float16* WQ  = (_Float16*)(ws + 14 * NBC);
    _Float16* WRK = WQ + W128;
    _Float16* WRV = WRK + W64;
    _Float16* WIK = WRV + W64;
    _Float16* WIV = WIK + W64;
    _Float16* WSR = WIV + W64;

    // Stage 0: pack all conv weights into WMMA A-fragment layout (f16).
    pack_weights<128><<<dim3((W128 + 255) / 256), dim3(256), 0, stream>>>(q_w, WQ);
    pack_weights<64><<<dim3((W64 + 255) / 256), dim3(256), 0, stream>>>(rk_w, WRK);
    pack_weights<64><<<dim3((W64 + 255) / 256), dim3(256), 0, stream>>>(rv_w, WRV);
    pack_weights<64><<<dim3((W64 + 255) / 256), dim3(256), 0, stream>>>(ik_w, WIK);
    pack_weights<64><<<dim3((W64 + 255) / 256), dim3(256), 0, stream>>>(iv_w, WIV);
    pack_weights<128><<<dim3((W128 + 255) / 256), dim3(256), 0, stream>>>(sr_w, WSR);

    const dim3 cgrid(256), cblk(128);
    // Stage 1: five conv+BN+ReLU layers -> f16 operands in WMMA-ready layouts.
    conv3_wmma<128, 0, true,  false><<<cgrid, cblk, 0, stream>>>(
        x, y, nullptr, WQ, q_s, q_b, Q16, nullptr);
    conv3_wmma<64,  0, false, false><<<cgrid, cblk, 0, stream>>>(
        x, nullptr, nullptr, WRK, rk_s, rk_b, RK16, nullptr);
    conv3_wmma<64,  1, false, false><<<cgrid, cblk, 0, stream>>>(
        x, nullptr, nullptr, WRV, rv_s, rv_b, RV16, nullptr);
    conv3_wmma<64,  0, false, false><<<cgrid, cblk, 0, stream>>>(
        y, nullptr, nullptr, WIK, ik_s, ik_b, IK16, nullptr);
    conv3_wmma<64,  1, false, false><<<cgrid, cblk, 0, stream>>>(
        y, nullptr, nullptr, WIV, iv_s, iv_b, IV16, nullptr);

    // Stage 2: two flash-attentions sharing Q; write the concat buffer.
    const dim3 agrid(64, 4), ablk(128);
    attend_wmma<<<agrid, ablk, 0, stream>>>(RK16, RV16, Q16, y, g1, REF, 0);
    attend_wmma<<<agrid, ablk, 0, stream>>>(IK16, IV16, Q16, x, g2, REF, 64);

    // Stage 3: final conv over the 128-ch concat -> fp32 output.
    conv3_wmma<128, 2, false, true><<<cgrid, cblk, 0, stream>>>(
        nullptr, nullptr, REF, WSR, sr_s, sr_b, nullptr, (float*)d_out);
}
